// MxDNADeforambleConvBlock_15513421873291
// MI455X (gfx1250) — matrix operand order repaired and checked by reading, not compiled, not run on hardware
//
#include <hip/hip_runtime.h>

#define B_   4
#define L_   4096
#define C_   768
#define KH_  7
#define KC_  (C_*KH_)        // 5376
#define PAD_ 3
#define KSTEPS_ (KC_/32)     // 168
#define LDSTR_  48           // LDS row stride in halves (96B, 16B-aligned)

typedef __attribute__((ext_vector_type(16))) __bf16 v16bf;
typedef __attribute__((ext_vector_type(8)))  __bf16 v8bf;
typedef __attribute__((ext_vector_type(8)))  float  v8f;

// ---------------------------------------------------------------------------
// Kernel 0: repack reg_w (C,C,KH) f32 -> Abf[o*KC + k*C + c] bf16
// ---------------------------------------------------------------------------
__global__ __launch_bounds__(256) void repack_w_kernel(const float* __restrict__ reg_w,
                                                       __bf16* __restrict__ Abf) {
  int idx = blockIdx.x * 256 + threadIdx.x;
  if (idx >= C_ * KC_) return;
  int o = idx / KC_;
  int r = idx - o * KC_;
  int k = r / C_;
  int c = r - k * C_;
  Abf[idx] = (__bf16)reg_w[(o * C_ + c) * KH_ + k];
}

// ---------------------------------------------------------------------------
// Kernel 1: layernorm -> hbf[b,p,c] (bf16, ungated), gate[b,p]
// one block per (b,p), 256 threads, 3 channels/thread
// ---------------------------------------------------------------------------
__global__ __launch_bounds__(256) void ln_kernel(const float* __restrict__ x,
                                                 const float* __restrict__ amask,
                                                 const float* __restrict__ smask,
                                                 const float* __restrict__ lnw,
                                                 const float* __restrict__ lnb,
                                                 __bf16* __restrict__ hbf,
                                                 float* __restrict__ gate) {
  const int p = blockIdx.x, b = blockIdx.y, t = threadIdx.x;
  const size_t row = ((size_t)b * L_ + p) * C_;
  const float v0 = x[row + t], v1 = x[row + t + 256], v2 = x[row + t + 512];
  __shared__ float rs[256], rq[256];
  rs[t] = v0 + v1 + v2;
  rq[t] = v0 * v0 + v1 * v1 + v2 * v2;
  __syncthreads();
  for (int off = 128; off > 0; off >>= 1) {
    if (t < off) { rs[t] += rs[t + off]; rq[t] += rq[t + off]; }
    __syncthreads();
  }
  __shared__ float mu_s, rstd_s;
  if (t == 0) {
    float mu = rs[0] * (1.0f / C_);
    float var = rq[0] * (1.0f / C_) - mu * mu;
    mu_s = mu;
    rstd_s = rsqrtf(var + 1e-12f);
    gate[(size_t)b * L_ + p] = (1.0f - smask[(size_t)b * L_ + p]) * amask[(size_t)b * L_ + p];
  }
  __syncthreads();
  const float mu = mu_s, rv = rstd_s;
  hbf[row + t]       = (__bf16)((v0 - mu) * rv * lnw[t]       + lnb[t]);
  hbf[row + t + 256] = (__bf16)((v1 - mu) * rv * lnw[t + 256] + lnb[t + 256]);
  hbf[row + t + 512] = (__bf16)((v2 - mu) * rv * lnw[t + 512] + lnb[t + 512]);
}

// ---------------------------------------------------------------------------
// Kernel 2: off/mod convs + fold into (i0,i1,w0,w1) per (b,k,p)
// block covers 256 positions of one batch; thread t: c-quarter (t&3)*192,
// 4 positions (t>>2)+64*i; 14 accumulators; LDS atomic reduce.
// ---------------------------------------------------------------------------
__global__ __launch_bounds__(256) void offsets_kernel(const __bf16* __restrict__ hbf,
                                                      const float* __restrict__ gate,
                                                      const float* __restrict__ off_w,
                                                      const float* __restrict__ off_b,
                                                      const float* __restrict__ mod_w,
                                                      const float* __restrict__ mod_b,
                                                      int* __restrict__ i0a,
                                                      int* __restrict__ i1a,
                                                      float* __restrict__ w0a,
                                                      float* __restrict__ w1a) {
  const int b = blockIdx.y;
  const int pbase = blockIdx.x * 256;
  const int t = threadIdx.x;
  __shared__ float accs[256 * 14];
  for (int i = t; i < 256 * 14; i += 256) accs[i] = 0.0f;
  __syncthreads();

  const int quarter = t & 3;
  const int prow = t >> 2;          // 0..63
  const int cbeg = quarter * 192;
  const __bf16* hb = hbf + (size_t)b * L_ * C_;

  float acc[4][14];
#pragma unroll
  for (int i = 0; i < 4; ++i)
#pragma unroll
    for (int s = 0; s < 14; ++s) acc[i][s] = 0.0f;

  for (int c = cbeg; c < cbeg + 192; ++c) {
    for (int j = 0; j < KH_; ++j) {
      float wo[7], wm[7];
#pragma unroll
      for (int ko = 0; ko < 7; ++ko) {
        wo[ko] = off_w[(ko * C_ + c) * KH_ + j];
        wm[ko] = mod_w[(ko * C_ + c) * KH_ + j];
      }
#pragma unroll
      for (int i = 0; i < 4; ++i) {
        const int p = pbase + prow + 64 * i;
        const int srow = p + j - PAD_;
        float hv = 0.0f;
        if (srow >= 0 && srow < L_) hv = (float)hb[(size_t)srow * C_ + c];
#pragma unroll
        for (int ko = 0; ko < 7; ++ko) {
          acc[i][ko]     += hv * wo[ko];
          acc[i][7 + ko] += hv * wm[ko];
        }
      }
    }
  }
#pragma unroll
  for (int i = 0; i < 4; ++i) {
    const int pl = prow + 64 * i;
#pragma unroll
    for (int s = 0; s < 14; ++s) atomicAdd(&accs[pl * 14 + s], acc[i][s]);
  }
  __syncthreads();

  for (int idx = t; idx < 256 * 7; idx += 256) {
    const int pl = idx / 7;
    const int ko = idx - pl * 7;
    const int p = pbase + pl;
    const float offv = accs[pl * 14 + ko] + off_b[ko];
    const float modv = 2.0f / (1.0f + expf(-(accs[pl * 14 + 7 + ko] + mod_b[ko])));
    const float pos = (float)(p - PAD_ + ko) + offv;
    const float f = floorf(pos);
    const float tt = pos - f;
    const int fi = (int)f;
    const bool in0 = (fi >= 0) && (fi <= L_ - 1);
    const bool in1 = (fi >= -1) && (fi <= L_ - 2);
    int i0 = fi < 0 ? 0 : (fi > L_ - 1 ? L_ - 1 : fi);
    int i1 = (fi + 1) < 0 ? 0 : ((fi + 1) > L_ - 1 ? L_ - 1 : fi + 1);
    const float g0 = gate[(size_t)b * L_ + i0];
    const float g1 = gate[(size_t)b * L_ + i1];
    const size_t oidx = ((size_t)b * KH_ + ko) * L_ + p;
    i0a[oidx] = i0;
    i1a[oidx] = i1;
    w0a[oidx] = in0 ? (1.0f - tt) * modv * g0 : 0.0f;
    w1a[oidx] = in1 ? tt * modv * g1 : 0.0f;
  }
}

// ---------------------------------------------------------------------------
// Kernel 3: main GEMM via WMMA bf16.
// out[b,p,o] = sum_{k,c} Abf[o, k*C+c] * (h[b,i0,c]*w0 + h[b,i1,c]*w1) + residual
// Per WG: 128(M=o) x 128(N=p); 8 waves, each 2x4 subtiles of 16x16.
// ---------------------------------------------------------------------------
__device__ __forceinline__ v16bf frag_a(const __bf16* rowp, int lsel) {
  // A 16x32 bf16 layout: lane<16: K0..7 then K16..23 ; lane>=16: K8..15 then K24..31
  v8bf lo = *(const v8bf*)(rowp + lsel * 8);
  v8bf hi = *(const v8bf*)(rowp + 16 + lsel * 8);
  v16bf r;
#pragma unroll
  for (int i = 0; i < 8; ++i) { r[i] = lo[i]; r[i + 8] = hi[i]; }
  return r;
}
__device__ __forceinline__ v16bf frag_b(const __bf16* rowp, int lsel) {
  // B 32x16 bf16 layout: lane<16: K0..7 then K16..23 ; lane>=16: K8..15 then K24..31
  v8bf lo = *(const v8bf*)(rowp + lsel * 8);
  v8bf hi = *(const v8bf*)(rowp + 16 + lsel * 8);
  v16bf r;
#pragma unroll
  for (int i = 0; i < 8; ++i) { r[i] = lo[i]; r[i + 8] = hi[i]; }
  return r;
}

__global__ __launch_bounds__(256) void gemm_kernel(const __bf16* __restrict__ Abf,
                                                   const __bf16* __restrict__ hbf,
                                                   const int* __restrict__ i0a,
                                                   const int* __restrict__ i1a,
                                                   const float* __restrict__ w0a,
                                                   const float* __restrict__ w1a,
                                                   const float* __restrict__ residual,
                                                   float* __restrict__ out) {
  const int nt = blockIdx.x, mt = blockIdx.y, b = blockIdx.z;
  const int P0 = nt * 128, M0 = mt * 128;
  const int t = threadIdx.x;

  __shared__ __bf16 ldsA[128 * LDSTR_];
  __shared__ __bf16 ldsS[128 * LDSTR_];

  const int wave = t >> 5, lane = t & 31;
  const int mblk = (wave & 3) * 32;
  const int nblk = (wave >> 2) * 64;
  const int lsel = lane >> 4;
  const int lrow = lane & 15;

  v8f acc[2][4];
#pragma unroll
  for (int mi = 0; mi < 2; ++mi)
#pragma unroll
    for (int ni = 0; ni < 4; ++ni)
#pragma unroll
      for (int r = 0; r < 8; ++r) acc[mi][ni][r] = 0.0f;

  const int rowA = t >> 1;
  const int segA = (t & 1) * 16;
  const int pS = t >> 1;
  const int csegS = (t & 1) * 16;

  int k = 0, c0 = 0;
  for (int ks = 0; ks < KSTEPS_; ++ks) {
    // ---- stage A slice: 128 x 32 bf16
    {
      const __bf16* ga = Abf + (size_t)(M0 + rowA) * KC_ + k * C_ + c0 + segA;
      *(v8bf*)&ldsA[rowA * LDSTR_ + segA]     = *(const v8bf*)ga;
      *(v8bf*)&ldsA[rowA * LDSTR_ + segA + 8] = *(const v8bf*)(ga + 8);
    }
    // ---- build sampled S slice: 128(p) x 32(c) bf16 with interp folded
    {
      const size_t base = ((size_t)b * KH_ + k) * L_ + P0 + pS;
      const int i0v = i0a[base];
      const int i1v = i1a[base];
      const float w0v = w0a[base];
      const float w1v = w1a[base];
      const v8bf* r0 = (const v8bf*)(hbf + ((size_t)b * L_ + i0v) * C_ + c0 + csegS);
      const v8bf* r1 = (const v8bf*)(hbf + ((size_t)b * L_ + i1v) * C_ + c0 + csegS);
      v8bf x0a = r0[0], x0b = r0[1];
      v8bf x1a = r1[0], x1b = r1[1];
      __bf16* sd = &ldsS[pS * LDSTR_ + csegS];
#pragma unroll
      for (int i = 0; i < 8; ++i)
        sd[i] = (__bf16)((float)x0a[i] * w0v + (float)x1a[i] * w1v);
#pragma unroll
      for (int i = 0; i < 8; ++i)
        sd[8 + i] = (__bf16)((float)x0b[i] * w0v + (float)x1b[i] * w1v);
    }
    __syncthreads();

    v16bf af[2], bfr[4];
#pragma unroll
    for (int mi = 0; mi < 2; ++mi)
      af[mi] = frag_a(&ldsA[(mblk + mi * 16 + lrow) * LDSTR_], lsel);
#pragma unroll
    for (int ni = 0; ni < 4; ++ni)
      bfr[ni] = frag_b(&ldsS[(nblk + ni * 16 + lrow) * LDSTR_], lsel);

#pragma unroll
    for (int mi = 0; mi < 2; ++mi)
#pragma unroll
      for (int ni = 0; ni < 4; ++ni)
        acc[mi][ni] = __builtin_amdgcn_wmma_f32_16x16x32_bf16(
            false, af[mi], false, bfr[ni], (short)0, acc[mi][ni], false, false);

    __syncthreads();
    c0 += 32;
    if (c0 == C_) { c0 = 0; ++k; }
  }

  // ---- epilogue: D(16x16 f32): lane<16 -> M=r, lane>=16 -> M=8+r ; N=lane&15
#pragma unroll
  for (int mi = 0; mi < 2; ++mi) {
#pragma unroll
    for (int ni = 0; ni < 4; ++ni) {
      const int p = P0 + nblk + ni * 16 + lrow;
      const int obase = M0 + mblk + mi * 16 + lsel * 8;
      const size_t oidx = ((size_t)b * L_ + p) * C_ + obase;
      const float4* rr = (const float4*)(residual + oidx);
      float4 r0 = rr[0], r1 = rr[1];
      v8f a = acc[mi][ni];
      float4 o0, o1;
      o0.x = a[0] + r0.x; o0.y = a[1] + r0.y; o0.z = a[2] + r0.z; o0.w = a[3] + r0.w;
      o1.x = a[4] + r1.x; o1.y = a[5] + r1.y; o1.z = a[6] + r1.z; o1.w = a[7] + r1.w;
      float4* oo = (float4*)(out + oidx);
      oo[0] = o0;
      oo[1] = o1;
    }
  }
}

// ---------------------------------------------------------------------------
extern "C" void kernel_launch(void* const* d_in, const int* in_sizes, int n_in,
                              void* d_out, int out_size, void* d_ws, size_t ws_size,
                              hipStream_t stream) {
  const float* hs    = (const float*)d_in[0];
  const float* amask = (const float*)d_in[1];
  const float* smask = (const float*)d_in[2];
  const float* lnw   = (const float*)d_in[3];
  const float* lnb   = (const float*)d_in[4];
  const float* off_w = (const float*)d_in[5];
  const float* off_b = (const float*)d_in[6];
  const float* mod_w = (const float*)d_in[7];
  const float* mod_b = (const float*)d_in[8];
  const float* reg_w = (const float*)d_in[9];
  float* out = (float*)d_out;

  char* ws = (char*)d_ws;
  __bf16* Abf = (__bf16*)ws;  ws += (size_t)C_ * KC_ * 2;          // 8,257,536
  __bf16* hbf = (__bf16*)ws;  ws += (size_t)B_ * L_ * C_ * 2;      // 25,165,824
  float* gate = (float*)ws;   ws += (size_t)B_ * L_ * 4;           // 65,536
  int*   i0a  = (int*)ws;     ws += (size_t)B_ * KH_ * L_ * 4;     // 458,752
  int*   i1a  = (int*)ws;     ws += (size_t)B_ * KH_ * L_ * 4;
  float* w0a  = (float*)ws;   ws += (size_t)B_ * KH_ * L_ * 4;
  float* w1a  = (float*)ws;   ws += (size_t)B_ * KH_ * L_ * 4;

  repack_w_kernel<<<dim3((C_ * KC_ + 255) / 256), dim3(256), 0, stream>>>(reg_w, Abf);
  ln_kernel<<<dim3(L_, B_), dim3(256), 0, stream>>>(hs, amask, smask, lnw, lnb, hbf, gate);
  offsets_kernel<<<dim3(L_ / 256, B_), dim3(256), 0, stream>>>(
      hbf, gate, off_w, off_b, mod_w, mod_b, i0a, i1a, w0a, w1a);
  gemm_kernel<<<dim3(L_ / 128, C_ / 128, B_), dim3(256), 0, stream>>>(
      Abf, hbf, i0a, i1a, w0a, w1a, hs, out);
}
